// GPT_41592463294722
// MI455X (gfx1250) — matrix operand order repaired and checked by reading, not compiled, not run on hardware
//
#include <hip/hip_runtime.h>
#include <math.h>

#define Bsz 4
#define Tn  2048
#define Cn  256
#define Hn  4
#define HSn 64
#define Ln  2
#define Vn  1024
#define BT  (Bsz*Tn)
#define BHn (Bsz*Hn)

typedef __attribute__((ext_vector_type(16))) __bf16 v16bf;
typedef __attribute__((ext_vector_type(8)))  float  v8f;
typedef unsigned int u32x4 __attribute__((ext_vector_type(4)));
typedef int          i32x4 __attribute__((ext_vector_type(4)));
typedef int          i32x8 __attribute__((ext_vector_type(8)));

#if defined(__has_builtin)
#if __has_builtin(__builtin_amdgcn_tensor_load_to_lds) && __has_builtin(__builtin_amdgcn_s_wait_tensorcnt)
#define HAVE_TDM 1
#endif
#endif
#ifndef HAVE_TDM
#define HAVE_TDM 0
#endif

#define LPAD 40  // padded LDS row pitch in bf16 elements (80B: 16B-aligned, conflict-free)

// ---------- bf16 helpers ----------
__device__ __forceinline__ __bf16 f2bf(float f) {
  union { float f; unsigned u; } x; x.f = f;
  unsigned r = x.u + 0x7FFFu + ((x.u >> 16) & 1u);
  unsigned short h = (unsigned short)(r >> 16);
  return __builtin_bit_cast(__bf16, h);
}
__device__ __forceinline__ float bf2f(__bf16 b) {
  unsigned short h = __builtin_bit_cast(unsigned short, b);
  union { unsigned u; float f; } x; x.u = ((unsigned)h) << 16;
  return x.f;
}
__device__ __forceinline__ float gelu_f(float x) {
  const float k = 0.7978845608028654f;
  return 0.5f * x * (1.0f + tanhf(k * (x + 0.044715f * x * x * x)));
}

// ---------- WMMA fragment loaders (bf16 16x16x32, wave32, ISA 7.12.2 layouts) ----------
__device__ __forceinline__ v16bf load_frag_a(const __bf16* __restrict__ A, int lda,
                                             int m0, int k0, int lane) {
  int m  = m0 + (lane & 15);
  int hi = lane >> 4;
  const __bf16* p = A + (size_t)m * lda + k0;
  union { v16bf v; uint4 q[2]; } u;
  u.q[0] = *(const uint4*)(p + 8 * hi);
  u.q[1] = *(const uint4*)(p + 16 + 8 * hi);
  return u.v;
}
__device__ __forceinline__ v16bf load_frag_b(const __bf16* __restrict__ W, int ldw,
                                             int n0, int k0, int lane) {
  int n  = n0 + (lane & 15);
  int hi = lane >> 4;
  const __bf16* p = W + (size_t)n * ldw + k0;
  union { v16bf v; uint4 q[2]; } u;
  u.q[0] = *(const uint4*)(p + 8 * hi);
  u.q[1] = *(const uint4*)(p + 16 + 8 * hi);
  return u.v;
}
// LDS variants: padded pitch LPAD, k-offset within the 32-wide staged tile
__device__ __forceinline__ v16bf lds_frag_a(const __bf16* T, int row0, int lane) {
  int hi = lane >> 4;
  const __bf16* p = T + (size_t)(row0 + (lane & 15)) * LPAD;
  union { v16bf v; uint4 q[2]; } u;
  u.q[0] = *(const uint4*)(p + 8 * hi);
  u.q[1] = *(const uint4*)(p + 16 + 8 * hi);
  return u.v;
}
__device__ __forceinline__ v16bf lds_frag_b(const __bf16* T, int row0, int lane) {
  int hi = lane >> 4;
  const __bf16* p = T + (size_t)(row0 + (lane & 15)) * LPAD;
  union { v16bf v; uint4 q[2]; } u;
  u.q[0] = *(const uint4*)(p + 8 * hi);
  u.q[1] = *(const uint4*)(p + 16 + 8 * hi);
  return u.v;
}
__device__ __forceinline__ v8f wmma_bf16(v16bf a, v16bf b, v8f c) {
  return __builtin_amdgcn_wmma_f32_16x16x32_bf16(false, a, false, b, (short)0, c, false, false);
}
__device__ __forceinline__ v8f zero_v8f() {
  v8f z;
#pragma unroll
  for (int i = 0; i < 8; ++i) z[i] = 0.0f;
  return z;
}

// ---------- TDM: DMA one (rows x 32) bf16 tile into LDS with 16B/row padding ----------
#if HAVE_TDM
__device__ __forceinline__ void tdm_load_tile(const __bf16* gptr, unsigned ldsOff,
                                              int rows, int ldElems) {
  unsigned long long ga = (unsigned long long)(uintptr_t)gptr;
  u32x4 g0;
  g0[0] = 1u;                                        // count=1, user descriptor
  g0[1] = ldsOff;                                    // lds_addr (bytes)
  g0[2] = (unsigned)(ga & 0xFFFFFFFFu);              // global_addr[31:0]
  g0[3] = (unsigned)((ga >> 32) & 0x01FFFFFFu) | 0x80000000u;  // addr[56:32] | type=2
  i32x8 g1;
  // data_size=2B(01<<16) | pad_enable(1<<20) | pad_interval=16dw(3<<22) | pad_amount=4dw(3<<25)
  g1[0] = (int)0x06D10000;
  g1[1] = (int)(((1u << 20) & 0xFFFFu) << 16);       // tensor_dim0[15:0]<<16  (dim0 = 1<<20)
  g1[2] = (int)(((1u << 20) >> 16) | ((((1u << 20)) & 0xFFFFu) << 16)); // dim0 hi | dim1 lo
  g1[3] = (int)(((1u << 20) >> 16) | (32u << 16));   // dim1 hi | tile_dim0=32
  g1[4] = rows;                                      // tile_dim1=rows, tile_dim2=0
  g1[5] = ldElems;                                   // tensor_dim0_stride[31:0]
  g1[6] = 0;
  g1[7] = 0;
  i32x4 z4; z4[0] = z4[1] = z4[2] = z4[3] = 0;
#if __clang_major__ >= 23
  i32x8 z8;
#pragma unroll
  for (int i = 0; i < 8; ++i) z8[i] = 0;
  __builtin_amdgcn_tensor_load_to_lds(g0, g1, z4, z4, z8, 0);
#else
  __builtin_amdgcn_tensor_load_to_lds(g0, g1, z4, z4, 0);
#endif
}
#endif

// ---------- elementwise / setup kernels ----------
__global__ void convert_f32_bf16(const float* __restrict__ in, __bf16* __restrict__ out, long n) {
  long i = (long)blockIdx.x * 256 + threadIdx.x;
  if (i < n) out[i] = f2bf(in[i]);
}

__global__ void transpose_conv_kernel(const float* __restrict__ in /*[T,HS]*/,
                                      __bf16* __restrict__ out /*[HS,T]*/) {
  long i = (long)blockIdx.x * 256 + threadIdx.x;
  if (i >= (long)Tn * HSn) return;
  int t = (int)(i / HSn), d = (int)(i % HSn);
  out[(size_t)d * Tn + t] = f2bf(in[i]);
}

__global__ void embed_kernel(const int* __restrict__ tokens, const float* __restrict__ wte,
                             float* __restrict__ x) {
  long i = (long)blockIdx.x * 256 + threadIdx.x;
  if (i >= (long)BT * Cn) return;
  long bt = i / Cn; int c = (int)(i % Cn);
  int tok = tokens[bt];
  x[i] = wte[(size_t)tok * Cn + c];
}

__global__ void ln_kernel(const float* __restrict__ x, const float* __restrict__ g,
                          const float* __restrict__ bta, __bf16* __restrict__ out) {
  __shared__ float sbuf[Cn];
  int row = blockIdx.x, tid = threadIdx.x;
  float v = x[(size_t)row * Cn + tid];
  sbuf[tid] = v; __syncthreads();
  for (int s = 128; s > 0; s >>= 1) { if (tid < s) sbuf[tid] += sbuf[tid + s]; __syncthreads(); }
  float mean = sbuf[0] * (1.0f / Cn); __syncthreads();
  float d = v - mean;
  sbuf[tid] = d * d; __syncthreads();
  for (int s = 128; s > 0; s >>= 1) { if (tid < s) sbuf[tid] += sbuf[tid + s]; __syncthreads(); }
  float var = sbuf[0] * (1.0f / Cn); __syncthreads();
  float r = rsqrtf(var + 1e-5f);
  out[(size_t)row * Cn + tid] = f2bf(d * r * g[tid] + bta[tid]);
}

__global__ void split_qkv_kernel(const float* __restrict__ qkv,
                                 __bf16* __restrict__ q, __bf16* __restrict__ k,
                                 __bf16* __restrict__ vT) {
  long i = (long)blockIdx.x * 256 + threadIdx.x;
  if (i >= (long)BT * 3 * Cn) return;
  int c3 = (int)(i % (3 * Cn)); long bt = i / (3 * Cn);
  int b = (int)(bt / Tn), t = (int)(bt % Tn);
  int which = c3 / Cn, cc = c3 % Cn;
  int h = cc / HSn, d = cc % HSn;
  __bf16 val = f2bf(qkv[i]);
  size_t bh = (size_t)b * Hn + h;
  if (which == 0)      q [(bh * Tn + t) * HSn + d] = val;
  else if (which == 1) k [(bh * Tn + t) * HSn + d] = val;
  else                 vT[(bh * HSn + d) * Tn + t] = val;
}

// ---------- dense GEMM: 128x64 block tile, TDM-staged LDS, double-buffered ----------
// out[m,n] = act(A@W^T + bias) (+residual); A bf16 [M,K] lda, W bf16 [N,K] ldw.
// Grid: (N/64, M/128), block 256 (8 waves). Wave w: rows m0=blk*128+w*16, 4 n-tiles.
__global__ void gemm_kernel(const __bf16* __restrict__ A, int lda,
                            const __bf16* __restrict__ W, int ldw,
                            const float* __restrict__ bias,
                            const float* __restrict__ residual,
                            float* __restrict__ outF, __bf16* __restrict__ outB,
                            int ldo, int M, int N, int K, int act) {
  __shared__ __bf16 ldsA[2][128 * LPAD];
  __shared__ __bf16 ldsB[2][64 * LPAD];
  int lane = threadIdx.x & 31, wave = threadIdx.x >> 5;
  int n0blk = blockIdx.x * 64;
  int m0blk = blockIdx.y * 128;
  int nk = K / 32;

#if HAVE_TDM
  if (wave == 0) {
    tdm_load_tile(A + (size_t)m0blk * lda, (unsigned)(uintptr_t)&ldsA[0][0], 128, lda);
    tdm_load_tile(W + (size_t)n0blk * ldw, (unsigned)(uintptr_t)&ldsB[0][0], 64, ldw);
  }
#endif

  v8f acc[4];
#pragma unroll
  for (int nt = 0; nt < 4; ++nt) acc[nt] = zero_v8f();

  for (int i = 0; i < nk; ++i) {
    int buf = i & 1;
#if HAVE_TDM
    if (wave == 0) {
      if (i + 1 < nk) {
        int k0n = (i + 1) * 32;
        tdm_load_tile(A + (size_t)m0blk * lda + k0n, (unsigned)(uintptr_t)&ldsA[buf ^ 1][0], 128, lda);
        tdm_load_tile(W + (size_t)n0blk * ldw + k0n, (unsigned)(uintptr_t)&ldsB[buf ^ 1][0], 64, ldw);
        __builtin_amdgcn_s_wait_tensorcnt(2);   // current pair retired, next pair in flight
      } else {
        __builtin_amdgcn_s_wait_tensorcnt(0);
      }
    }
    __syncthreads();
#else
    {  // cooperative staging fallback
      int k0 = i * 32;
      for (int e = threadIdx.x; e < (128 + 64) * 4; e += 256) {
        int row, q; const __bf16* src; __bf16* dst;
        if (e < 128 * 4) {
          row = e >> 2; q = e & 3;
          src = A + (size_t)(m0blk + row) * lda + k0 + q * 8;
          dst = &ldsA[buf][row * LPAD + q * 8];
        } else {
          int e2 = e - 128 * 4; row = e2 >> 2; q = e2 & 3;
          src = W + (size_t)(n0blk + row) * ldw + k0 + q * 8;
          dst = &ldsB[buf][row * LPAD + q * 8];
        }
        *(uint4*)dst = *(const uint4*)src;
      }
    }
    __syncthreads();
#endif
    v16bf a = lds_frag_a(&ldsA[buf][0], wave * 16, lane);
#pragma unroll
    for (int nt = 0; nt < 4; ++nt) {
      v16bf b = lds_frag_b(&ldsB[buf][0], nt * 16, lane);
      acc[nt] = wmma_bf16(a, b, acc[nt]);
    }
    __syncthreads();
  }

  int hi = lane >> 4;
  int m0 = m0blk + wave * 16;
#pragma unroll
  for (int nt = 0; nt < 4; ++nt) {
    int n = n0blk + nt * 16 + (lane & 15);
    float bv = bias ? bias[n] : 0.0f;
#pragma unroll
    for (int i = 0; i < 8; ++i) {
      int m = m0 + 8 * hi + i;
      float val = acc[nt][i] + bv;
      if (act) val = gelu_f(val);
      size_t o = (size_t)m * ldo + n;
      if (residual) val += residual[o];
      if (outF) outF[o] = val;
      if (outB) outB[o] = f2bf(val);
    }
  }
}

// ---------- batched score GEMM: wave = 16x64 strip, lower-tri only ----------
__global__ void attn_score_kernel(const __bf16* __restrict__ Q,
                                  const __bf16* __restrict__ Wk, long wStride,
                                  __bf16* __restrict__ out) {
  int lane = threadIdx.x & 31, wave = threadIdx.x >> 5;
  int bh = blockIdx.z;
  int n0b = blockIdx.x * 64;
  int m0 = (blockIdx.y * 8 + wave) * 16;
  if (n0b > m0) return;                       // whole strip above diagonal (wave-uniform)
  const __bf16* Ab = Q  + (size_t)bh * Tn * HSn;
  const __bf16* Wb = Wk + (size_t)bh * wStride;
  v8f acc[4];
#pragma unroll
  for (int nt = 0; nt < 4; ++nt) acc[nt] = zero_v8f();
#pragma unroll
  for (int k0 = 0; k0 < HSn; k0 += 32) {
    v16bf a = load_frag_a(Ab, HSn, m0, k0, lane);
#pragma unroll
    for (int nt = 0; nt < 4; ++nt) {
      v16bf b = load_frag_b(Wb, HSn, n0b + nt * 16, k0, lane);
      acc[nt] = wmma_bf16(a, b, acc[nt]);
    }
  }
  int hi = lane >> 4;
  __bf16* obh = out + (size_t)bh * Tn * Tn;
#pragma unroll
  for (int nt = 0; nt < 4; ++nt) {
    int nt0 = n0b + nt * 16;
    if (nt0 <= m0) {                          // skip stores of fully-masked tiles
      int n = nt0 + (lane & 15);
#pragma unroll
      for (int i = 0; i < 8; ++i) {
        int m = m0 + 8 * hi + i;
        obh[(size_t)m * Tn + n] = f2bf(acc[nt][i]);
      }
    }
  }
}

// ---------- fused mask + relative bias + softmax + skew ----------
__global__ void softmax_rel_kernel(const __bf16* __restrict__ S,
                                   const __bf16* __restrict__ QE,
                                   __bf16* __restrict__ att, __bf16* __restrict__ A2) {
  __shared__ float sbuf[256];
  int t = blockIdx.x, bh = blockIdx.y, tid = threadIdx.x;
  size_t rowoff = ((size_t)bh * Tn + t) * Tn;
  const float scale = 0.125f;                 // 1/sqrt(HS)
  float vals[Tn / 256];
  float mx = -3.0e38f;
#pragma unroll
  for (int j = 0; j < Tn / 256; ++j) {
    int s = j * 256 + tid;
    float v;
    if (s <= t) v = scale * bf2f(S[rowoff + s]) + bf2f(QE[rowoff + (t - s)]);
    else        v = -__builtin_inff();
    vals[j] = v; mx = fmaxf(mx, v);
  }
  sbuf[tid] = mx; __syncthreads();
  for (int s = 128; s > 0; s >>= 1) { if (tid < s) sbuf[tid] = fmaxf(sbuf[tid], sbuf[tid + s]); __syncthreads(); }
  mx = sbuf[0]; __syncthreads();
  float sum = 0.0f;
#pragma unroll
  for (int j = 0; j < Tn / 256; ++j) { float e = __expf(vals[j] - mx); vals[j] = e; sum += e; }
  sbuf[tid] = sum; __syncthreads();
  for (int s = 128; s > 0; s >>= 1) { if (tid < s) sbuf[tid] += sbuf[tid + s]; __syncthreads(); }
  float inv = 1.0f / sbuf[0]; __syncthreads();
#pragma unroll
  for (int j = 0; j < Tn / 256; ++j) {
    int s = j * 256 + tid;
    float a = vals[j] * inv;
    att[rowoff + s] = f2bf(a);
    if (s <= t) A2[rowoff + (t - s)] = f2bf(a);
    else        A2[rowoff + s]       = f2bf(0.0f);
  }
}

// ---------- dual GEMM: y = att@v + A2@embv ; wave = 16x64 strip, K causally bounded ----------
__global__ void attn_out_kernel(const __bf16* __restrict__ att,
                                const __bf16* __restrict__ A2m,
                                const __bf16* __restrict__ vT,    // [BH,HS,T]
                                const __bf16* __restrict__ embvT, // [HS,T]
                                __bf16* __restrict__ yc) {        // [B,T,C]
  int lane = threadIdx.x & 31, wave = threadIdx.x >> 5;
  int bh = blockIdx.y;
  int m0 = (blockIdx.x * 8 + wave) * 16;
  const __bf16* A1b = att + (size_t)bh * Tn * Tn;
  const __bf16* A2b = A2m + (size_t)bh * Tn * Tn;
  const __bf16* W1b = vT  + (size_t)bh * HSn * Tn;
  v8f acc[4];
#pragma unroll
  for (int nt = 0; nt < 4; ++nt) acc[nt] = zero_v8f();
  int kend = m0 + 16;                          // rows beyond t contribute zero
  for (int k0 = 0; k0 < kend; k0 += 32) {
    v16bf a1 = load_frag_a(A1b, Tn, m0, k0, lane);
    v16bf a2 = load_frag_a(A2b, Tn, m0, k0, lane);
#pragma unroll
    for (int nt = 0; nt < 4; ++nt) {
      v16bf b1 = load_frag_b(W1b,   Tn, nt * 16, k0, lane);
      v16bf b2 = load_frag_b(embvT, Tn, nt * 16, k0, lane);
      acc[nt] = wmma_bf16(a1, b1, acc[nt]);
      acc[nt] = wmma_bf16(a2, b2, acc[nt]);
    }
  }
  int b = bh / Hn, h = bh % Hn;
  int hi = lane >> 4;
#pragma unroll
  for (int nt = 0; nt < 4; ++nt) {
    int n = nt * 16 + (lane & 15);
#pragma unroll
    for (int i = 0; i < 8; ++i) {
      int m = m0 + 8 * hi + i;
      yc[((size_t)(b * Tn + m)) * Cn + h * HSn + n] = f2bf(acc[nt][i]);
    }
  }
}

// ---------------------------------------------------------------------------
extern "C" void kernel_launch(void* const* d_in, const int* in_sizes, int n_in,
                              void* d_out, int out_size, void* d_ws, size_t ws_size,
                              hipStream_t stream) {
  (void)in_sizes; (void)n_in; (void)out_size; (void)ws_size;
  const int*   tokens  = (const int*)  d_in[0];
  const float* wte     = (const float*)d_in[1];
  const float* ln1_g   = (const float*)d_in[2];
  const float* ln1_b   = (const float*)d_in[3];
  const float* attn_w  = (const float*)d_in[4];
  const float* attn_b  = (const float*)d_in[5];
  const float* proj_w  = (const float*)d_in[6];
  const float* proj_b  = (const float*)d_in[7];
  const float* embk    = (const float*)d_in[8];
  const float* embv    = (const float*)d_in[9];
  const float* ln2_g   = (const float*)d_in[10];
  const float* ln2_b   = (const float*)d_in[11];
  const float* fc_w    = (const float*)d_in[12];
  const float* fc_b    = (const float*)d_in[13];
  const float* mproj_w = (const float*)d_in[14];
  const float* mproj_b = (const float*)d_in[15];
  const float* lnf_g   = (const float*)d_in[16];
  const float* lnf_b   = (const float*)d_in[17];
  const float* head_w  = (const float*)d_in[18];
  float* logits = (float*)d_out;

  char* ws = (char*)d_ws;
  size_t off = 0;
  auto alloc = [&](size_t bytes) -> char* {
    size_t o = off; off += (bytes + 255) & ~(size_t)255; return ws + o;
  };

  float*  x     = (float*) alloc((size_t)BT * Cn * 4);
  __bf16* hbuf  = (__bf16*)alloc((size_t)BT * Cn * 2);
  float*  qkv   = (float*) alloc((size_t)BT * 3 * Cn * 4);
  __bf16* qb    = (__bf16*)alloc((size_t)BT * Cn * 2);
  __bf16* kb    = (__bf16*)alloc((size_t)BT * Cn * 2);
  __bf16* vTb   = (__bf16*)alloc((size_t)BT * Cn * 2);
  __bf16* Sbuf  = (__bf16*)alloc((size_t)BHn * Tn * Tn * 2);
  __bf16* QEbuf = (__bf16*)alloc((size_t)BHn * Tn * Tn * 2);
  __bf16* attb  = (__bf16*)alloc((size_t)BHn * Tn * Tn * 2);
  __bf16* A2b   = (__bf16*)alloc((size_t)BHn * Tn * Tn * 2);
  __bf16* yc    = (__bf16*)alloc((size_t)BT * Cn * 2);
  __bf16* fcact = (__bf16*)alloc((size_t)BT * 4 * Cn * 2);
  __bf16* attn_wb  = (__bf16*)alloc((size_t)Ln * 3 * Cn * Cn * 2);
  __bf16* proj_wb  = (__bf16*)alloc((size_t)Ln * Cn * Cn * 2);
  __bf16* fc_wb    = (__bf16*)alloc((size_t)Ln * 4 * Cn * Cn * 2);
  __bf16* mproj_wb = (__bf16*)alloc((size_t)Ln * Cn * 4 * Cn * 2);
  __bf16* head_wb  = (__bf16*)alloc((size_t)Vn * Cn * 2);
  __bf16* embk_b   = (__bf16*)alloc((size_t)Ln * Tn * HSn * 2);
  __bf16* embvT_b  = (__bf16*)alloc((size_t)Ln * HSn * Tn * 2);

  auto conv = [&](const float* in, __bf16* out, long n) {
    convert_f32_bf16<<<(unsigned)((n + 255) / 256), 256, 0, stream>>>(in, out, n);
  };
  conv(attn_w,  attn_wb,  (long)Ln * 3 * Cn * Cn);
  conv(proj_w,  proj_wb,  (long)Ln * Cn * Cn);
  conv(fc_w,    fc_wb,    (long)Ln * 4 * Cn * Cn);
  conv(mproj_w, mproj_wb, (long)Ln * Cn * 4 * Cn);
  conv(head_w,  head_wb,  (long)Vn * Cn);
  conv(embk,    embk_b,   (long)Ln * Tn * HSn);
  for (int l = 0; l < Ln; ++l)
    transpose_conv_kernel<<<(Tn * HSn + 255) / 256, 256, 0, stream>>>(
        embv + (size_t)l * Tn * HSn, embvT_b + (size_t)l * HSn * Tn);

  embed_kernel<<<((long)BT * Cn + 255) / 256, 256, 0, stream>>>(tokens, wte, x);

  auto gemm = [&](const __bf16* A, int lda, const __bf16* W, int ldw,
                  const float* bias, const float* residual,
                  float* outF, __bf16* outB, int ldo, int M, int N, int K, int act) {
    dim3 g(N / 64, M / 128);
    gemm_kernel<<<g, 256, 0, stream>>>(A, lda, W, ldw, bias, residual, outF, outB,
                                       ldo, M, N, K, act);
  };

  for (int l = 0; l < Ln; ++l) {
    // ---- attention ----
    ln_kernel<<<BT, Cn, 0, stream>>>(x, ln1_g + l * Cn, ln1_b + l * Cn, hbuf);
    gemm(hbuf, Cn, attn_wb + (size_t)l * 3 * Cn * Cn, Cn, attn_b + l * 3 * Cn,
         nullptr, qkv, nullptr, 3 * Cn, BT, 3 * Cn, Cn, 0);
    split_qkv_kernel<<<((long)BT * 3 * Cn + 255) / 256, 256, 0, stream>>>(qkv, qb, kb, vTb);

    dim3 gs(Tn / 64, Tn / 128, BHn);
    attn_score_kernel<<<gs, 256, 0, stream>>>(qb, kb, (long)Tn * HSn, Sbuf);          // S  = q.k^T
    attn_score_kernel<<<gs, 256, 0, stream>>>(qb, embk_b + (size_t)l * Tn * HSn, 0L,  // QE = q.embk^T
                                              QEbuf);
    softmax_rel_kernel<<<dim3(Tn, BHn), 256, 0, stream>>>(Sbuf, QEbuf, attb, A2b);

    attn_out_kernel<<<dim3(Tn / 128, BHn), 256, 0, stream>>>(
        attb, A2b, vTb, embvT_b + (size_t)l * HSn * Tn, yc);
    gemm(yc, Cn, proj_wb + (size_t)l * Cn * Cn, Cn, proj_b + l * Cn,
         x, x, nullptr, Cn, BT, Cn, Cn, 0);                                           // x += proj(y)
    // ---- MLP ----
    ln_kernel<<<BT, Cn, 0, stream>>>(x, ln2_g + l * Cn, ln2_b + l * Cn, hbuf);
    gemm(hbuf, Cn, fc_wb + (size_t)l * 4 * Cn * Cn, Cn, fc_b + l * 4 * Cn,
         nullptr, nullptr, fcact, 4 * Cn, BT, 4 * Cn, Cn, 1);                         // gelu -> bf16
    gemm(fcact, 4 * Cn, mproj_wb + (size_t)l * Cn * 4 * Cn, 4 * Cn, mproj_b + l * Cn,
         x, x, nullptr, Cn, BT, Cn, 4 * Cn, 0);                                       // x += mproj
  }

  ln_kernel<<<BT, Cn, 0, stream>>>(x, lnf_g, lnf_b, hbuf);
  gemm(hbuf, Cn, head_wb, Cn, nullptr, nullptr, logits, nullptr, Vn, BT, Vn, Cn, 0);
}
